// RelativeMultiheadSelfAttentionModule_65060164600163
// MI455X (gfx1250) — compile-verified, matrix operand order checked by reading, not run on hardware
//
#include <hip/hip_runtime.h>
#include <math.h>

#define B_  4
#define T_  2048
#define D_  512
#define H_  8
#define DH_ 64

typedef __bf16 bf16;
typedef __attribute__((ext_vector_type(16))) __bf16        v16bf;
typedef __attribute__((ext_vector_type(8)))  float         v8f;
typedef __attribute__((ext_vector_type(8)))  unsigned int  v8u;
typedef __attribute__((ext_vector_type(4)))  unsigned int  v4u;

union FragU { v8u u; v16bf h; };
union V8U  { uint4 q; bf16 e[8]; };

__device__ __forceinline__ bf16 to_bf16(float f) {
  unsigned int x = __builtin_bit_cast(unsigned int, f);
  x += 0x7FFFu + ((x >> 16) & 1u);                 // round-to-nearest-even
  unsigned short s = (unsigned short)(x >> 16);
  return __builtin_bit_cast(bf16, s);
}

// K offset for element-pair p of a 16-bit WMMA A/B fragment (ISA 7.12.2):
// lanes 0-15 hold K {0..7,16..23}, lanes 16-31 hold K {8..15,24..31}
__device__ __forceinline__ int koff(int p, int hi) {
  return (p < 4 ? 2 * p : 16 + 2 * (p - 4)) + 8 * hi;
}

__device__ __forceinline__ v16bf load_frag(const bf16* row, int hi) {
  FragU f;
#pragma unroll
  for (int p = 0; p < 8; ++p)
    f.u[p] = *reinterpret_cast<const unsigned int*>(row + koff(p, hi));
  return f.h;
}

__device__ __forceinline__ v16bf zero_frag() {
  FragU f;
#pragma unroll
  for (int p = 0; p < 8; ++p) f.u[p] = 0u;
  return f.h;
}

__device__ __forceinline__ v8f wmma_bf16(v16bf a, v16bf b, v8f c) {
  return __builtin_amdgcn_wmma_f32_16x16x32_bf16(false, a, false, b,
                                                 (short)0, c, false, false);
}

// TDM: DMA `nelem` contiguous bf16 elements from global -> LDS (1-D tile).
// Issued by one wave; caller must s_wait_tensorcnt + barrier.
__device__ __forceinline__ void tdm_load_1d(unsigned ldsAddr,
                                            unsigned long long gAddr,
                                            unsigned nelem) {
  v4u g0;
  g0[0] = 1u;                                   // count=1, user mode
  g0[1] = ldsAddr;                              // lds_addr
  g0[2] = (unsigned)(gAddr & 0xFFFFFFFFu);      // global_addr[31:0]
  g0[3] = (unsigned)((gAddr >> 32) & 0x01FFFFFFu) | (2u << 30); // [56:32] | type=2
  v8u g1;
  g1[0] = 1u << 16;                             // data_size = 2 bytes
  g1[1] = (nelem & 0xFFFFu) << 16;              // tensor_dim0[15:0]
  g1[2] = ((nelem >> 16) & 0xFFFFu) | (1u << 16); // tensor_dim0[31:16], tensor_dim1=1
  g1[3] = (nelem & 0xFFFFu) << 16;              // tile_dim0
  g1[4] = 1u;                                   // tile_dim1 = 1
  g1[5] = nelem;                                // tensor_dim0_stride lo
  g1[6] = (nelem & 0xFFFFu) << 16;              // tensor_dim1_stride lo
  g1[7] = 0u;
  asm volatile("tensor_load_to_lds %0, %1" :: "s"(g0), "s"(g1) : "memory");
}

// ---------------------------------------------------------------- utilities
__global__ void k_transpose_w(const float* __restrict__ src, bf16* __restrict__ dst) {
  int idx = blockIdx.x * blockDim.x + threadIdx.x;   // 512*512
  int n = idx >> 9, k = idx & 511;
  dst[idx] = to_bf16(src[k * 512 + n]);              // dst[n*512+k] = W[k][n]
}

__global__ void k_pe(bf16* __restrict__ pe) {
  int idx = blockIdx.x * blockDim.x + threadIdx.x;   // T * D/2
  int t = idx >> 8, i = idx & 255;
  float div = __expf(-logf(10000.f) * (float)(2 * i) / (float)D_);
  float ang = (float)t * div;
  pe[t * D_ + 2 * i]     = to_bf16(__sinf(ang));
  pe[t * D_ + 2 * i + 1] = to_bf16(__cosf(ang));
}

__global__ void __launch_bounds__(256)
k_layernorm(const float* __restrict__ Q, const float* __restrict__ g,
            const float* __restrict__ bb, bf16* __restrict__ qn,
            bf16* __restrict__ qraw) {
  __shared__ float red[8];
  const int row = blockIdx.x, tid = threadIdx.x;
  const float* src = Q + (size_t)row * D_;
  float x0 = src[tid], x1 = src[tid + 256];
  float s = x0 + x1;
#pragma unroll
  for (int off = 16; off >= 1; off >>= 1) s += __shfl_xor(s, off, 32);
  if ((tid & 31) == 0) red[tid >> 5] = s;
  __syncthreads();
  float tot = 0.f;
#pragma unroll
  for (int i = 0; i < 8; ++i) tot += red[i];
  float mu = tot * (1.f / (float)D_);
  float d0 = x0 - mu, d1 = x1 - mu;
  float v = d0 * d0 + d1 * d1;
#pragma unroll
  for (int off = 16; off >= 1; off >>= 1) v += __shfl_xor(v, off, 32);
  __syncthreads();
  if ((tid & 31) == 0) red[tid >> 5] = v;
  __syncthreads();
  float vt = 0.f;
#pragma unroll
  for (int i = 0; i < 8; ++i) vt += red[i];
  float inv = rsqrtf(vt * (1.f / (float)D_) + 1e-5f);
  bf16* qnr = qn + (size_t)row * D_;
  bf16* qrr = qraw + (size_t)row * D_;
  qnr[tid]       = to_bf16(d0 * inv * g[tid] + bb[tid]);
  qnr[tid + 256] = to_bf16(d1 * inv * g[tid + 256] + bb[tid + 256]);
  qrr[tid]       = to_bf16(x0);
  qrr[tid + 256] = to_bf16(x1);
}

// ---------------------------------------------------------------- GEMM
enum { MODE_QUV = 0, MODE_BH = 1, MODE_POS = 2, MODE_OUT = 3 };

__global__ void __launch_bounds__(256)
k_gemm(const bf16* __restrict__ A, const bf16* __restrict__ Wt,
       const float* __restrict__ bias, int mode,
       bf16* __restrict__ o0, bf16* __restrict__ o1,
       const float* __restrict__ ub, const float* __restrict__ vb,
       float* __restrict__ of) {
  __shared__ __align__(16) bf16 As[16 * 512];        // 16KB A strip
  const int tid = threadIdx.x;
  const size_t m0 = (size_t)blockIdx.x * 16;
  // Tensor Data Mover pulls the contiguous 16KB A strip into LDS (wave 0).
  if (tid < 32) {
    tdm_load_1d((unsigned)(size_t)(&As[0]),
                (unsigned long long)(size_t)(A + m0 * 512), 16 * 512);
    __builtin_amdgcn_s_wait_tensorcnt(0);
  }
  __syncthreads();
  const int w = tid >> 5, lane = tid & 31, hi = lane >> 4, c = lane & 15;
  const int n0 = blockIdx.y * 128 + w * 16;
  const bf16* wrow = Wt + (size_t)(n0 + c) * 512;
  v8f acc = {};
#pragma unroll 4
  for (int kk = 0; kk < 16; ++kk) {
    v16bf af = load_frag(As + c * 512 + kk * 32, hi);   // row m = lane&15
    v16bf bf_ = load_frag(wrow + kk * 32, hi);          // col n = n0 + (lane&15)
    acc = wmma_bf16(af, bf_, acc);
  }
#pragma unroll
  for (int r = 0; r < 8; ++r) {
    int mg = (int)m0 + r + 8 * hi;
    int ng = n0 + c;
    float val = acc[r] + (bias ? bias[ng] : 0.f);
    if (mode == MODE_OUT) {
      of[(size_t)mg * 512 + ng] = val;
    } else if (mode == MODE_POS) {
      o0[((size_t)(ng >> 6) * T_ + mg) * DH_ + (ng & 63)] = to_bf16(val);
    } else {
      int bbat = mg >> 11, t = mg & (T_ - 1);
      int hh = ng >> 6, d = ng & 63;
      size_t oi = (((size_t)bbat * H_ + hh) * T_ + t) * DH_ + d;
      if (mode == MODE_BH) {
        o0[oi] = to_bf16(val);
      } else {                                         // MODE_QUV
        o0[oi] = to_bf16(val + ub[ng]);
        o1[oi] = to_bf16(val + vb[ng]);
      }
    }
  }
}

// ------------------------------------------------- flash relative attention
__global__ void __launch_bounds__(256)
k_attn(const bf16* __restrict__ qu, const bf16* __restrict__ qv,
       const bf16* __restrict__ kk_, const bf16* __restrict__ vv,
       const bf16* __restrict__ pos, const unsigned char* __restrict__ mask,
       bf16* __restrict__ ctx) {
  __shared__ __align__(16) bf16 Ks[32 * 64];   // K tile [key][d]
  __shared__ __align__(16) bf16 Vt[64 * 32];   // V tile transposed [d][key]
  __shared__ __align__(16) bf16 Pb[8][16 * 32];// per-wave P relayout buffer
  const int tid = threadIdx.x;
  const int w = tid >> 5, lane = tid & 31, hi = lane >> 4, c = lane & 15;
  const int bh = blockIdx.y, b = bh >> 3, h = bh & 7;
  const int tq0 = blockIdx.x * 128 + w * 16;
  const bf16* quh = qu + ((size_t)bh * T_ + tq0) * DH_;
  const bf16* qvh = qv + ((size_t)bh * T_ + tq0) * DH_;
  const bf16* kh  = kk_ + (size_t)bh * T_ * DH_;
  const bf16* vh  = vv  + (size_t)bh * T_ * DH_;
  const bf16* ph  = pos + (size_t)h * T_ * DH_;
  const unsigned char* mrow = mask + (size_t)b * T_;
  const unsigned KsBase = (unsigned)(size_t)(&Ks[0]);

  v16bf quf[2], qvf[2];
#pragma unroll
  for (int ks = 0; ks < 2; ++ks) {
    quf[ks] = load_frag(quh + c * DH_ + ks * 32, hi);
    qvf[ks] = load_frag(qvh + c * DH_ + ks * 32, hi);
  }
  v8f O[4] = {};
  float mst[8], lst[8];
#pragma unroll
  for (int r = 0; r < 8; ++r) { mst[r] = -1e30f; lst[r] = 0.f; }
  const float scale = 0.044194173824159216f;   // 1/sqrt(512)

  for (int kb = 0; kb < T_ / 32; ++kb) {
    const int tk0 = kb * 32;
    __syncthreads();
    {  // K tile: async copy straight into LDS (16B/lane, no VGPR round-trip)
      unsigned long long ga =
          (unsigned long long)(size_t)(kh + (size_t)tk0 * DH_) +
          (unsigned long long)tid * 16ull;
      unsigned la = KsBase + (unsigned)tid * 16u;
      asm volatile("global_load_async_to_lds_b128 %0, %1, off"
                   :: "v"(la), "v"(ga) : "memory");
      // V tile: VGPR path (transposed on the way into LDS)
      int key = tid >> 3, d0 = (tid & 7) * 8;
      V8U vq;
      vq.q = *reinterpret_cast<const uint4*>(vh + ((size_t)tk0 + key) * DH_ + d0);
#pragma unroll
      for (int e = 0; e < 8; ++e) Vt[(d0 + e) * 32 + key] = vq.e[e];
    }
    if (kb + 1 < T_ / 32)
      __builtin_prefetch(kh + (size_t)(tk0 + 32) * DH_ + tid * 8, 0, 0);
    asm volatile("s_wait_asynccnt 0" ::: "memory");
    __syncthreads();

    v8f S[2];
#pragma unroll
    for (int si = 0; si < 2; ++si) {
      v8f s_ = {};
#pragma unroll
      for (int ks = 0; ks < 2; ++ks) {               // content: qu @ k^T
        v16bf bk_ = load_frag(Ks + (si * 16 + c) * DH_ + ks * 32, hi);
        s_ = wmma_bf16(quf[ks], bk_, s_);
      }
      // pos score band x[t, base..base+31], then rel_shift gather
      const int base = T_ - 16 - tq0 + tk0 + si * 16;
      v8f xA = {}, xB = {};
#pragma unroll
      for (int ks = 0; ks < 2; ++ks) {
        int jA = base + c, jB = base + 16 + c;
        v16bf ba = (jA < T_) ? load_frag(ph + (size_t)jA * DH_ + ks * 32, hi)
                             : zero_frag();
        v16bf bbf = (jB < T_) ? load_frag(ph + (size_t)jB * DH_ + ks * 32, hi)
                              : zero_frag();
        xA = wmma_bf16(qvf[ks], ba, xA);
        xB = wmma_bf16(qvf[ks], bbf, xB);
      }
#pragma unroll
      for (int r = 0; r < 8; ++r) {                  // rel_shift: j = base + 15 - m + c
        int idx = 15 - r - 8 * hi + c;
        float a  = __shfl(xA[r], (idx & 15) + (hi << 4), 32);
        float b2 = __shfl(xB[r], ((idx - 16) & 15) + (hi << 4), 32);
        s_[r] += (idx < 16) ? a : b2;
      }
      unsigned char mk = mrow[tk0 + si * 16 + c];
#pragma unroll
      for (int r = 0; r < 8; ++r)
        s_[r] = mk ? s_[r] * scale : -1e9f;
      S[si] = s_;
    }
    // online softmax (rows live across the 16-lane half-wave)
#pragma unroll
    for (int r = 0; r < 8; ++r) {
      float mx = fmaxf(S[0][r], S[1][r]);
#pragma unroll
      for (int off = 1; off < 16; off <<= 1) mx = fmaxf(mx, __shfl_xor(mx, off, 32));
      float mnew = fmaxf(mst[r], mx);
      float sc = __expf(mst[r] - mnew);
      mst[r] = mnew;
      float p0 = __expf(S[0][r] - mnew);
      float p1 = __expf(S[1][r] - mnew);
      S[0][r] = p0; S[1][r] = p1;
      float rs = p0 + p1;
#pragma unroll
      for (int off = 1; off < 16; off <<= 1) rs += __shfl_xor(rs, off, 32);
      lst[r] = lst[r] * sc + rs;
#pragma unroll
      for (int f = 0; f < 4; ++f) O[f][r] *= sc;
    }
    // re-layout P (C frag) -> A frag via per-wave LDS (in-order DS, no barrier)
    bf16* pb = &Pb[w][0];
#pragma unroll
    for (int r = 0; r < 8; ++r) {
      int rowm = r + 8 * hi;
      pb[rowm * 32 + c]      = to_bf16(S[0][r]);
      pb[rowm * 32 + 16 + c] = to_bf16(S[1][r]);
    }
    v16bf pf = load_frag(pb + c * 32, hi);
#pragma unroll
    for (int f = 0; f < 4; ++f) {                    // O += P @ V  (K = 32 keys)
      v16bf bv = load_frag(Vt + (f * 16 + c) * 32, hi);
      O[f] = wmma_bf16(pf, bv, O[f]);
    }
  }
#pragma unroll
  for (int r = 0; r < 8; ++r) {
    float inv = 1.f / lst[r];
    int t = tq0 + r + 8 * hi;
    bf16* crow = ctx + ((size_t)b * T_ + t) * D_ + h * DH_;
#pragma unroll
    for (int f = 0; f < 4; ++f)
      crow[f * 16 + c] = to_bf16(O[f][r] * inv);
  }
}

// ---------------------------------------------------------------- launcher
extern "C" void kernel_launch(void* const* d_in, const int* in_sizes, int n_in,
                              void* d_out, int out_size, void* d_ws, size_t ws_size,
                              hipStream_t stream) {
  (void)in_sizes; (void)n_in; (void)out_size; (void)ws_size;
  const float* Q    = (const float*)d_in[0];
  const unsigned char* mask = (const unsigned char*)d_in[1];
  const float* ln_g = (const float*)d_in[2];
  const float* ln_b = (const float*)d_in[3];
  const float* Wq   = (const float*)d_in[4];
  const float* bq   = (const float*)d_in[5];
  const float* Wk   = (const float*)d_in[6];
  const float* bk   = (const float*)d_in[7];
  const float* Wv   = (const float*)d_in[8];
  const float* bv   = (const float*)d_in[9];
  const float* Wpos = (const float*)d_in[10];
  const float* ub   = (const float*)d_in[11];
  const float* vb   = (const float*)d_in[12];
  const float* Wo   = (const float*)d_in[13];
  const float* bo   = (const float*)d_in[14];
  float* out = (float*)d_out;

  char* ws = (char*)d_ws;
  size_t off = 0;
  auto alloc = [&](size_t bytes) {
    char* p = ws + off;
    off += (bytes + 255) & ~(size_t)255;
    return p;
  };
  const size_t WB = (size_t)512 * 512 * sizeof(bf16);
  const size_t XB = (size_t)B_ * T_ * D_ * sizeof(bf16);
  bf16* WqT  = (bf16*)alloc(WB);
  bf16* WkT  = (bf16*)alloc(WB);
  bf16* WvT  = (bf16*)alloc(WB);
  bf16* WpT  = (bf16*)alloc(WB);
  bf16* WoT  = (bf16*)alloc(WB);
  bf16* peb  = (bf16*)alloc((size_t)T_ * D_ * sizeof(bf16));
  bf16* qnb  = (bf16*)alloc(XB);
  bf16* qrb  = (bf16*)alloc(XB);
  bf16* qub  = (bf16*)alloc(XB);
  bf16* qvb  = (bf16*)alloc(XB);
  bf16* kkb  = (bf16*)alloc(XB);
  bf16* vvb  = (bf16*)alloc(XB);
  bf16* posb = (bf16*)alloc((size_t)T_ * D_ * sizeof(bf16));
  bf16* ctxb = (bf16*)alloc(XB);

  k_transpose_w<<<1024, 256, 0, stream>>>(Wq, WqT);
  k_transpose_w<<<1024, 256, 0, stream>>>(Wk, WkT);
  k_transpose_w<<<1024, 256, 0, stream>>>(Wv, WvT);
  k_transpose_w<<<1024, 256, 0, stream>>>(Wpos, WpT);
  k_transpose_w<<<1024, 256, 0, stream>>>(Wo, WoT);
  k_pe<<<(T_ * (D_ / 2)) / 256, 256, 0, stream>>>(peb);
  k_layernorm<<<B_ * T_, 256, 0, stream>>>(Q, ln_g, ln_b, qnb, qrb);

  dim3 gg(B_ * T_ / 16, 4);
  k_gemm<<<gg, 256, 0, stream>>>(qnb, WqT, bq, MODE_QUV, qub, qvb, ub, vb, nullptr);
  k_gemm<<<gg, 256, 0, stream>>>(qrb, WkT, bk, MODE_BH, kkb, nullptr, nullptr, nullptr, nullptr);
  k_gemm<<<gg, 256, 0, stream>>>(qrb, WvT, bv, MODE_BH, vvb, nullptr, nullptr, nullptr, nullptr);
  dim3 gp(T_ / 16, 4);
  k_gemm<<<gp, 256, 0, stream>>>(peb, WpT, nullptr, MODE_POS, posb, nullptr, nullptr, nullptr, nullptr);

  dim3 ga(T_ / 128, B_ * H_);
  k_attn<<<ga, 256, 0, stream>>>(qub, qvb, kkb, vvb, posb, mask, ctxb);

  k_gemm<<<gg, 256, 0, stream>>>(ctxb, WoT, bo, MODE_OUT, nullptr, nullptr, nullptr, nullptr, out);
}
